// GraphSage3TPK_48155173323147
// MI455X (gfx1250) — matrix operand order repaired and checked by reading, not compiled, not run on hardware
//
#include <hip/hip_runtime.h>
#include <hip/hip_bf16.h>
#include <math.h>

typedef __attribute__((ext_vector_type(16))) __bf16 v16bf;
typedef __attribute__((ext_vector_type(8)))  __bf16 v8bf;
typedef __attribute__((ext_vector_type(8)))  float  v8f;

#define HDIM 128
#define NGRAPH 128
#define NPG 1024
#define NCLS 10

__device__ __forceinline__ unsigned short f2bf(float f) {
    unsigned int u = __float_as_uint(f);
    unsigned int r = u + 0x7FFFu + ((u >> 16) & 1u);   // round-to-nearest-even
    return (unsigned short)(r >> 16);
}

// ---------------------------------------------------------------------------
// Pack [Wl | Wr] (each HxH fp32, row-major, out-major) into per-lane
// WMMA B fragments (bf16, contiguous k order, same as A), per layer. B[k][n] = Wcat[n][k], K=256, N=128.
// Element index within a layer: (((c*8 + s)*32 + lane)*16 + j)
//   n = c*16 + (lane&15), k = s*32 + (lane>>4)*16 + j
// ---------------------------------------------------------------------------
__global__ void pack_w_kernel(const float* __restrict__ Wl1, const float* __restrict__ Wr1,
                              const float* __restrict__ Wl2, const float* __restrict__ Wr2,
                              const float* __restrict__ Wl3, const float* __restrict__ Wr3,
                              unsigned short* __restrict__ pw) {
    int tid = blockIdx.x * blockDim.x + threadIdx.x;
    if (tid >= 3 * 32768) return;
    int layer = tid >> 15;
    int r = tid & 32767;
    int j    = r & 15;
    int lane = (r >> 4) & 31;
    int s    = (r >> 9) & 7;
    int c    = (r >> 12) & 7;
    int n = c * 16 + (lane & 15);
    int k = s * 32 + (lane >> 4) * 16 + j;
    const float* Wl = (layer == 0) ? Wl1 : (layer == 1) ? Wl2 : Wl3;
    const float* Wr = (layer == 0) ? Wr1 : (layer == 1) ? Wr2 : Wr3;
    float w = (k < HDIM) ? Wl[n * HDIM + k] : Wr[n * HDIM + (k - HDIM)];
    pw[tid] = f2bf(w);
}

__global__ void init_nm_kernel(int* __restrict__ nm, int N) {
    int t = blockIdx.x * blockDim.x + threadIdx.x;
    if (t < N) nm[t] = 1;
}

__global__ void zero_sd_kernel(float* __restrict__ s, float* __restrict__ deg, int N) {
    long long tid = (long long)blockIdx.x * blockDim.x + threadIdx.x;
    long long tot = (long long)N * HDIM;
    if (tid < tot)            s[tid] = 0.f;
    else if (tid < tot + N)   deg[tid - tot] = 0.f;
}

// ---------------------------------------------------------------------------
// Edge scatter: one wave per edge; 32 lanes x float4 atomicAdd of x[src] into
// s[dst]; lane0 bumps deg[dst]. Edge alive iff both endpoints alive.
// ---------------------------------------------------------------------------
__global__ void agg_kernel(const float* __restrict__ x,
                           const int* __restrict__ src, const int* __restrict__ dst,
                           const int* __restrict__ nm,
                           float* __restrict__ s, float* __restrict__ deg, int E) {
    int w = (blockIdx.x * blockDim.x + threadIdx.x) >> 5;
    int lane = threadIdx.x & 31;
    if (w >= E) return;
    int si = src[w], di = dst[w];
    if (!(nm[si] & nm[di])) return;
    const float4 m = *(const float4*)(x + (long long)si * HDIM + lane * 4);
    float* out = s + (long long)di * HDIM + lane * 4;
    atomicAdd(out + 0, m.x);
    atomicAdd(out + 1, m.y);
    atomicAdd(out + 2, m.z);
    atomicAdd(out + 3, m.w);
    if (lane == 0) atomicAdd(deg + di, 1.0f);
}

// ---------------------------------------------------------------------------
// Fused SAGE linear: y = relu( [s/deg | x] @ [Wl|Wr]^T + bl ), K=256.
// One block = 8 row tiles (128 rows); wave ct = one 16-wide output column.
// B fragments preloaded into registers once, reused across 8 row tiles.
// Inner loop: 2x v8f load -> v_pk_mul_f32 -> packed f32->bf16 cvt -> WMMA.
// ---------------------------------------------------------------------------
__global__ __launch_bounds__(256) void sage_gemm_kernel(
        const float* __restrict__ sbuf, const float* __restrict__ deg,
        const float* __restrict__ xin, const unsigned short* __restrict__ pw,
        const float* __restrict__ bl, float* __restrict__ y) {
    const int ct   = threadIdx.x >> 5;   // column tile 0..7
    const int lane = threadIdx.x & 31;
    const int half = lane >> 4;
    const int m    = lane & 15;

    // Preload this wave's 8 B fragments (256-K x 16-N column slab), reused 8x.
    v16bf bfr[8];
#pragma unroll
    for (int ks = 0; ks < 8; ++ks)
        bfr[ks] = *(const v16bf*)(pw + ((size_t)(ct * 8 + ks) * 32 + lane) * 16);

    const int n  = lane & 15;
    const int jc = ct * 16 + n;
    const float bias = bl[jc];

    for (int rt = 0; rt < 8; ++rt) {
        const long long row_tile = (long long)blockIdx.x * 8 + rt;
        const long long row = row_tile * 16 + m;
        const float inv = 1.0f / fmaxf(deg[row], 1.0f);
        const float* srow = sbuf + row * HDIM;
        const float* xrow = xin  + row * HDIM;

        v8f acc = {0.f, 0.f, 0.f, 0.f, 0.f, 0.f, 0.f, 0.f};

#pragma unroll
        for (int ks = 0; ks < 8; ++ks) {
            const float* ap = (ks < 4) ? srow : xrow;
            const float  sc = (ks < 4) ? inv  : 1.0f;
            const int k0 = (ks & 3) * 32 + half * 16;
            v8f f0 = *(const v8f*)(ap + k0);        // 32B aligned
            v8f f1 = *(const v8f*)(ap + k0 + 8);    // 32B aligned
            f0 = f0 * sc;
            f1 = f1 * sc;
            union { v16bf v; v8bf h[2]; } a;
            a.h[0] = __builtin_convertvector(f0, v8bf);
            a.h[1] = __builtin_convertvector(f1, v8bf);
            acc = __builtin_amdgcn_wmma_f32_16x16x32_bf16(
                /*neg_a=*/false, a.v, /*neg_b=*/false, bfr[ks],
                /*c_mod=*/(short)0, acc, /*reuse_a=*/false, /*reuse_b=*/false);
        }

#pragma unroll
        for (int r = 0; r < 8; ++r) {
            long long mr = row_tile * 16 + r + 8 * half;
            float v = acc[r] + bias;
            y[mr * HDIM + jc] = v > 0.f ? v : 0.f;
        }
    }
}

// ---------------------------------------------------------------------------
// TopK pooling per graph: score = x.w/||w||, keep top kk among alive nodes,
// gate kept nodes by tanh(score), zero the rest, update node mask.
// ---------------------------------------------------------------------------
__global__ __launch_bounds__(1024) void topk_kernel(float* __restrict__ x,
                                                    const float* __restrict__ wp,
                                                    int* __restrict__ nm, int kk) {
    __shared__ float swp[HDIM];
    __shared__ float red[HDIM];
    __shared__ unsigned int scnt;
    __shared__ int pre[NPG];
    const int t = threadIdx.x;
    const int g = blockIdx.x;

    if (t < HDIM) { float w = wp[t]; swp[t] = w; red[t] = w * w; }
    __syncthreads();
    for (int s = 64; s > 0; s >>= 1) {
        if (t < s) red[t] += red[t + s];
        __syncthreads();
    }
    const float norm = sqrtf(red[0]) + 1e-16f;

    const long long i = (long long)g * NPG + t;
    const float* xr = x + i * HDIM;
    float sc = 0.f;
    for (int f = 0; f < HDIM; ++f) sc += xr[f] * swp[f];
    sc /= norm;

    const int alive = nm[i];
    unsigned int u;
    {
        unsigned int b = __float_as_uint(sc);
        u = (b & 0x80000000u) ? ~b : (b | 0x80000000u);
    }
    if (!alive) u = 0u;  // masked nodes rank lowest (-inf in reference)

    unsigned int T = 0;
    for (int bit = 31; bit >= 0; --bit) {
        unsigned int cand = T | (1u << bit);
        if (t == 0) scnt = 0;
        __syncthreads();
        if (u >= cand) atomicAdd(&scnt, 1u);
        __syncthreads();
        if (scnt >= (unsigned)kk) T = cand;
        __syncthreads();
    }
    if (t == 0) scnt = 0;
    __syncthreads();
    if (u > T) atomicAdd(&scnt, 1u);
    __syncthreads();
    const int cntGT = (int)scnt;

    const int flag = (u == T) ? 1 : 0;
    pre[t] = flag;
    __syncthreads();
    for (int off = 1; off < NPG; off <<= 1) {
        int v = (t >= off) ? pre[t - off] : 0;
        __syncthreads();
        pre[t] += v;
        __syncthreads();
    }
    const int excl = pre[t] - flag;
    const int sel = alive && ((u > T) || (flag && excl < (kk - cntGT)));

    const float gate = tanhf(sc);
    float* xw = x + i * HDIM;
    for (int f = 0; f < HDIM; ++f) xw[f] = sel ? xw[f] * gate : 0.f;
    nm[i] = sel;
}

// ---------------------------------------------------------------------------
// Per-graph: mean-pool over alive nodes (pruned rows are exactly zero),
// then relu(pooled@Wf1^T+bf1), logits, log_softmax.
// ---------------------------------------------------------------------------
__global__ __launch_bounds__(128) void final_kernel(const float* __restrict__ x,
                                                    const int* __restrict__ nm,
                                                    const float* __restrict__ Wf1,
                                                    const float* __restrict__ bf1,
                                                    const float* __restrict__ Wf2,
                                                    const float* __restrict__ bf2,
                                                    float* __restrict__ out) {
    __shared__ float pooled[HDIM];
    __shared__ float hbuf[64];
    __shared__ float lg[NCLS];
    __shared__ int cred[HDIM];
    __shared__ float mmax, lse;
    const int t = threadIdx.x;
    const int g = blockIdx.x;

    float sum = 0.f;
    const float* base = x + (long long)g * NPG * HDIM;
    for (int node = 0; node < NPG; ++node) sum += base[(long long)node * HDIM + t];
    int cnt = 0;
    for (int node = t; node < NPG; node += HDIM) cnt += nm[(long long)g * NPG + node];
    cred[t] = cnt;
    __syncthreads();
    for (int s = 64; s > 0; s >>= 1) {
        if (t < s) cred[t] += cred[t + s];
        __syncthreads();
    }
    pooled[t] = sum / fmaxf((float)cred[0], 1.0f);
    __syncthreads();

    if (t < 64) {
        float acc = bf1[t];
        for (int f = 0; f < HDIM; ++f) acc += pooled[f] * Wf1[t * HDIM + f];
        hbuf[t] = acc > 0.f ? acc : 0.f;
    }
    __syncthreads();
    if (t < NCLS) {
        float acc = bf2[t];
        for (int j = 0; j < 64; ++j) acc += hbuf[j] * Wf2[t * 64 + j];
        lg[t] = acc;
    }
    __syncthreads();
    if (t == 0) {
        float mx = lg[0];
        for (int c = 1; c < NCLS; ++c) mx = fmaxf(mx, lg[c]);
        float ss = 0.f;
        for (int c = 0; c < NCLS; ++c) ss += expf(lg[c] - mx);
        mmax = mx; lse = logf(ss);
    }
    __syncthreads();
    if (t < NCLS) out[g * NCLS + t] = lg[t] - mmax - lse;
}

// ---------------------------------------------------------------------------
extern "C" void kernel_launch(void* const* d_in, const int* in_sizes, int n_in,
                              void* d_out, int out_size, void* d_ws, size_t ws_size,
                              hipStream_t stream) {
    const float* x   = (const float*)d_in[0];
    const int*   ei  = (const int*)d_in[1];
    const float* Wl1 = (const float*)d_in[2];
    const float* bl1 = (const float*)d_in[3];
    const float* Wr1 = (const float*)d_in[4];
    const float* wp1 = (const float*)d_in[5];
    const float* Wl2 = (const float*)d_in[6];
    const float* bl2 = (const float*)d_in[7];
    const float* Wr2 = (const float*)d_in[8];
    const float* wp2 = (const float*)d_in[9];
    const float* Wl3 = (const float*)d_in[10];
    const float* bl3 = (const float*)d_in[11];
    const float* Wr3 = (const float*)d_in[12];
    const float* wp3 = (const float*)d_in[13];
    const float* Wf1 = (const float*)d_in[14];
    const float* bf1 = (const float*)d_in[15];
    const float* Wf2 = (const float*)d_in[16];
    const float* bf2 = (const float*)d_in[17];

    const int N = in_sizes[0] / HDIM;      // 131072
    const int E = in_sizes[1] / 2;         // 2097152
    const int* srcp = ei;
    const int* dstp = ei + E;

    char* ws = (char*)d_ws;
    size_t off = 0;
    float* sbuf = (float*)(ws + off); off += (size_t)N * HDIM * 4;
    float* bufA = (float*)(ws + off); off += (size_t)N * HDIM * 4;
    float* bufB = (float*)(ws + off); off += (size_t)N * HDIM * 4;
    float* deg  = (float*)(ws + off); off += (size_t)N * 4;
    int*   nm   = (int*)(ws + off);   off += (size_t)N * 4;
    unsigned short* pw = (unsigned short*)(ws + off);

    const int kks[3] = {820, 656, 525};
    const float* bls[3] = {bl1, bl2, bl3};
    const float* wps[3] = {wp1, wp2, wp3};

    pack_w_kernel<<<(3 * 32768 + 255) / 256, 256, 0, stream>>>(Wl1, Wr1, Wl2, Wr2, Wl3, Wr3, pw);
    init_nm_kernel<<<(N + 255) / 256, 256, 0, stream>>>(nm, N);

    const long long zelems = (long long)N * (HDIM + 1);
    const unsigned zblocks = (unsigned)((zelems + 255) / 256);
    const unsigned ablocks = (unsigned)(((long long)E * 32 + 255) / 256);

    const float* xin = x;
    float* ybufs[3] = {bufA, bufB, bufA};
    for (int l = 0; l < 3; ++l) {
        float* y = ybufs[l];
        zero_sd_kernel<<<zblocks, 256, 0, stream>>>(sbuf, deg, N);
        agg_kernel<<<ablocks, 256, 0, stream>>>(xin, srcp, dstp, nm, sbuf, deg, E);
        sage_gemm_kernel<<<N / 128, 256, 0, stream>>>(sbuf, deg, xin, pw + (size_t)l * 32768, bls[l], y);
        topk_kernel<<<NGRAPH, NPG, 0, stream>>>(y, wps[l], nm, kks[l]);
        xin = y;
    }

    final_kernel<<<NGRAPH, HDIM, 0, stream>>>(bufA, nm, Wf1, bf1, Wf2, bf2, (float*)d_out);
}
